// BiLSTM_CRF_30562987278496
// MI455X (gfx1250) — compile-verified, matrix operand order checked by reading, not run on hardware
//
#include <hip/hip_runtime.h>

typedef __attribute__((ext_vector_type(16))) __bf16 v16bf;
typedef __attribute__((ext_vector_type(8)))  __bf16 v8bf;
typedef __attribute__((ext_vector_type(8)))  float  v8f;

namespace {
constexpr int S   = 4096;   // sentence length
constexpr int E   = 300;    // embedding dim
constexpr int EP  = 320;    // embedding dim padded to mult of 32 (WMMA K)
constexpr int HH  = 256;    // per-direction hidden
constexpr int G4  = 1024;   // 4*HH gate rows
constexpr int H   = 512;    // concat hidden
constexpr int T   = 24;     // tags
constexpr int TP  = 32;     // tags padded to WMMA tile
constexpr int START_TAG = 22;
constexpr int STOP_TAG  = 23;
constexpr float NEG = -10000.0f;
}

// ---------------------------------------------------------------------------
// Embedding gather + f32->bf16 cast, K-padded to 320.
// ---------------------------------------------------------------------------
__global__ void k_embed(const int* __restrict__ sent,
                        const float* __restrict__ emb,
                        __bf16* __restrict__ x) {
  const int s = blockIdx.x;          // one block per token
  const int t = threadIdx.x;         // blockDim = 320
  float v = (t < E) ? emb[(long)sent[s] * E + t] : 0.0f;
  x[s * EP + t] = (__bf16)v;
}

// ---------------------------------------------------------------------------
// Generic f32 -> bf16 cast with zero padding (rows and cols).
// ---------------------------------------------------------------------------
__global__ void k_castpad(const float* __restrict__ src, __bf16* __restrict__ dst,
                          int srows, int scols, int drows, int dcols) {
  int i = blockIdx.x * blockDim.x + threadIdx.x;
  if (i >= drows * dcols) return;
  int r = i / dcols, c = i % dcols;
  float v = (r < srows && c < scols) ? src[r * scols + c] : 0.0f;
  dst[i] = (__bf16)v;
}

// ---------------------------------------------------------------------------
// C[M,N] = A[M,K] * B[N,K]^T + bias[n], bf16 inputs, f32 accum via
// v_wmma_f32_16x16x32_bf16. One wave per 16x16 output tile.
// A-frag: lane L holds row m0+L%16, K-chunks (L/16)*8+{0..7} and +16.
// B-frag (B = W^T, so col n of B = row n of W): identical addressing on W.
// D layout: lane (n%16)+16*(m/8), vgpr r -> m = r + 8*(lane/16).
// ---------------------------------------------------------------------------
__global__ void k_gemm_wmma(const __bf16* __restrict__ A, int lda,
                            const __bf16* __restrict__ W, int ldw,
                            const float*  __restrict__ bias, int nbias,
                            float* __restrict__ C, int ldc,
                            int mtiles, int ntiles, int K) {
  const int wave = blockIdx.x * (blockDim.x >> 5) + (threadIdx.x >> 5);
  if (wave >= mtiles * ntiles) return;           // wave-uniform: EXEC stays all-1
  const int tm   = wave / ntiles;
  const int tn   = wave % ntiles;
  const int lane = threadIdx.x & 31;
  const int half = lane >> 4;                    // 0: K 0-7/16-23, 1: K 8-15/24-31
  const int row  = lane & 15;

  const __bf16* ap = A + (long)(tm * 16 + row) * lda + half * 8;
  const __bf16* wp = W + (long)(tn * 16 + row) * ldw + half * 8;

  v8f acc = {};
  for (int k0 = 0; k0 < K; k0 += 32) {
    v8bf alo = *(const v8bf*)(ap + k0);
    v8bf ahi = *(const v8bf*)(ap + k0 + 16);
    v8bf blo = *(const v8bf*)(wp + k0);
    v8bf bhi = *(const v8bf*)(wp + k0 + 16);
    v16bf a, b;
#pragma unroll
    for (int i = 0; i < 8; ++i) {
      a[i] = alo[i]; a[i + 8] = ahi[i];
      b[i] = blo[i]; b[i + 8] = bhi[i];
    }
    acc = __builtin_amdgcn_wmma_f32_16x16x32_bf16(
        /*neg_a=*/false, a, /*neg_b=*/false, b,
        /*c_mod=*/(short)0, acc, /*reuse_a=*/false, /*reuse_b=*/false);
  }

  const int n  = tn * 16 + row;
  const float bv = (n < nbias) ? bias[n] : 0.0f;
#pragma unroll
  for (int r = 0; r < 8; ++r) {
    const int m = tm * 16 + r + half * 8;
    C[(long)m * ldc + n] = acc[r] + bv;
  }
}

// ---------------------------------------------------------------------------
// Sequential LSTM recurrence, one 1024-thread block per direction.
// pre[t][g] already holds x_t . W_ih^T + b (from WMMA GEMM).
// Thread g computes gate row g: pre + dot(W_hh[g,:], h_prev). W_hh stays
// L2-resident (1 MB/dir << 192 MB); h_prev lives in LDS.
// PyTorch gate order i,f,g,o. Writes bf16 h into hout[t][dir*256 + u].
// ---------------------------------------------------------------------------
__global__ void k_lstm(const float* __restrict__ pre_f,
                       const float* __restrict__ pre_b,
                       const float* __restrict__ whh_f,
                       const float* __restrict__ whh_b,
                       __bf16* __restrict__ hout) {
  const int dir = blockIdx.x;
  const float* __restrict__ pre = dir ? pre_b : pre_f;
  const float* __restrict__ whh = dir ? whh_b : whh_f;

  __shared__ float hprev[HH];
  __shared__ float gbuf[G4];
  const int tid = threadIdx.x;   // 1024
  if (tid < HH) hprev[tid] = 0.0f;
  float c = 0.0f;
  __syncthreads();

  const float4* __restrict__ wr = (const float4*)(whh + (long)tid * HH);
  for (int t = 0; t < S; ++t) {
    const int ti = dir ? (S - 1 - t) : t;
    float acc = pre[(long)ti * G4 + tid];
    const float4* hp = (const float4*)hprev;
#pragma unroll 8
    for (int j = 0; j < HH / 4; ++j) {
      float4 w = wr[j], h4 = hp[j];
      acc += w.x * h4.x + w.y * h4.y + w.z * h4.z + w.w * h4.w;
    }
    gbuf[tid] = acc;
    __syncthreads();
    if (tid < HH) {
      const float ig = 1.0f / (1.0f + __expf(-gbuf[tid]));
      const float fg = 1.0f / (1.0f + __expf(-gbuf[HH + tid]));
      const float gg = tanhf(gbuf[2 * HH + tid]);
      const float og = 1.0f / (1.0f + __expf(-gbuf[3 * HH + tid]));
      c = fg * c + ig * gg;
      const float h = og * tanhf(c);
      hprev[tid] = h;
      hout[(long)ti * H + dir * HH + tid] = (__bf16)h;
    }
    __syncthreads();
  }
}

// ---------------------------------------------------------------------------
// Viterbi decode on a single wave32. Transitions cached in LDS; lane "to"
// scans all "from" with strict > (matches jnp.argmax first-max). Lane 0 does
// STOP termination + backtrace. out[0]=score, out[1..S]=path (as floats).
// ---------------------------------------------------------------------------
__global__ void k_viterbi(const float* __restrict__ feats,
                          const float* __restrict__ trans,
                          int* __restrict__ bptr,
                          float* __restrict__ out) {
  __shared__ float tl[T * T];
  __shared__ float v[T];
  __shared__ float vn[T];
  const int tid = threadIdx.x;   // 32
  for (int i = tid; i < T * T; i += 32) tl[i] = trans[i];
  if (tid < T) v[tid] = (tid == START_TAG) ? 0.0f : NEG;
  __syncthreads();

  for (int t = 0; t < S; ++t) {
    if (tid < T) {
      float best = -3.4e38f; int arg = 0;
#pragma unroll
      for (int f = 0; f < T; ++f) {
        const float s = v[f] + tl[tid * T + f];   // trans[to][from] + v[from]
        if (s > best) { best = s; arg = f; }
      }
      vn[tid] = best + feats[t * TP + tid];
      bptr[t * T + tid] = arg;
    }
    __syncthreads();
    if (tid < T) v[tid] = vn[tid];
    __syncthreads();
  }

  if (tid == 0) {
    float best = -3.4e38f; int bt = 0;
    for (int f = 0; f < T; ++f) {
      const float s = v[f] + tl[STOP_TAG * T + f];
      if (s > best) { best = s; bt = f; }
    }
    out[0] = best;
    int tag = bt;
    for (int t = S - 1; t >= 0; --t) {
      out[1 + t] = (float)tag;
      tag = bptr[t * T + tag];
    }
  }
}

// ---------------------------------------------------------------------------
extern "C" void kernel_launch(void* const* d_in, const int* in_sizes, int n_in,
                              void* d_out, int out_size, void* d_ws, size_t ws_size,
                              hipStream_t stream) {
  const int*   sentence = (const int*)  d_in[0];
  const float* emb      = (const float*)d_in[1];
  const float* w_ih_f   = (const float*)d_in[2];
  const float* w_hh_f   = (const float*)d_in[3];
  const float* b_f      = (const float*)d_in[4];
  const float* w_ih_b   = (const float*)d_in[5];
  const float* w_hh_b   = (const float*)d_in[6];
  const float* b_b      = (const float*)d_in[7];
  const float* w_tag    = (const float*)d_in[8];
  const float* b_tag    = (const float*)d_in[9];
  const float* trans    = (const float*)d_in[10];

  // Workspace carve-out (256B-aligned slabs), ~43 MB total.
  size_t off = 0;
  auto carve = [&](size_t bytes) -> void* {
    void* p = (char*)d_ws + off;
    off += (bytes + 255) & ~(size_t)255;
    return p;
  };
  __bf16* x_bf    = (__bf16*)carve((size_t)S * EP * 2);
  __bf16* wihf_bf = (__bf16*)carve((size_t)G4 * EP * 2);
  __bf16* wihb_bf = (__bf16*)carve((size_t)G4 * EP * 2);
  __bf16* wtag_bf = (__bf16*)carve((size_t)TP * H * 2);
  float*  pre_f   = (float*) carve((size_t)S * G4 * 4);
  float*  pre_b   = (float*) carve((size_t)S * G4 * 4);
  __bf16* h_bf    = (__bf16*)carve((size_t)S * H * 2);
  float*  feats   = (float*) carve((size_t)S * TP * 4);
  int*    bptrs   = (int*)   carve((size_t)S * T * 4);

  // 1) embedding gather (cast to bf16, pad K to 320)
  k_embed<<<S, EP, 0, stream>>>(sentence, emb, x_bf);

  // 2) weight casts/pads
  {
    int n = G4 * EP;
    k_castpad<<<(n + 255) / 256, 256, 0, stream>>>(w_ih_f, wihf_bf, G4, E, G4, EP);
    k_castpad<<<(n + 255) / 256, 256, 0, stream>>>(w_ih_b, wihb_bf, G4, E, G4, EP);
    int m = TP * H;
    k_castpad<<<(m + 255) / 256, 256, 0, stream>>>(w_tag, wtag_bf, T, H, TP, H);
  }

  // 3) time-parallel input projections via WMMA: pre = x . W_ih^T + b
  {
    const int mtiles = S / 16, ntiles = G4 / 16;           // 256 x 64
    const int waves  = mtiles * ntiles;
    const int blocks = (waves + 7) / 8;                    // 8 waves / 256-thd block
    k_gemm_wmma<<<blocks, 256, 0, stream>>>(x_bf, EP, wihf_bf, EP, b_f, G4,
                                            pre_f, G4, mtiles, ntiles, EP);
    k_gemm_wmma<<<blocks, 256, 0, stream>>>(x_bf, EP, wihb_bf, EP, b_b, G4,
                                            pre_b, G4, mtiles, ntiles, EP);
  }

  // 4) sequential bidirectional recurrence (2 WGPs in parallel)
  k_lstm<<<2, 1024, 0, stream>>>(pre_f, pre_b, w_hh_f, w_hh_b, h_bf);

  // 5) tag projection via WMMA: feats = h . W_tag^T + b_tag (N padded to 32)
  {
    const int mtiles = S / 16, ntiles = TP / 16;           // 256 x 2
    const int waves  = mtiles * ntiles;
    const int blocks = (waves + 7) / 8;
    k_gemm_wmma<<<blocks, 256, 0, stream>>>(h_bf, H, wtag_bf, H, b_tag, T,
                                            feats, TP, mtiles, ntiles, H);
  }

  // 6) Viterbi decode + backtrace
  k_viterbi<<<1, 32, 0, stream>>>(feats, trans, bptrs, (float*)d_out);
}
